// GAT_50156628082753
// MI455X (gfx1250) — matrix operand order repaired and checked by reading, not compiled, not run on hardware
//
#include <hip/hip_runtime.h>
#include <hip/hip_bf16.h>
#include <math.h>

#define NNODE 50000
#define NEDGE 800000
#define FDIM  128      // IN and hidden width (H*D = 128 for both layers)
#define NHEAD 4
#define HDIM  32
#define SLOPE 0.2f
#define MTPW  5        // M tiles per wave in the GEMM (3125 = 625*5)

typedef __attribute__((ext_vector_type(16))) __bf16 v16bf;
typedef __attribute__((ext_vector_type(8)))  float  v8f;

union ABu { v16bf v; unsigned short us[16]; uint4 q[2]; };

__device__ __forceinline__ unsigned short f2bf(float f) {
  unsigned int u = __float_as_uint(f);
  u += 0x7FFFu + ((u >> 16) & 1u);      // round-to-nearest-even
  return (unsigned short)(u >> 16);
}

__global__ void k_fill(float* __restrict__ p, float v, int n) {
  int i = blockIdx.x * blockDim.x + threadIdx.x;
  if (i < n) p[i] = v;
}

__global__ void k_conv_bf16(const float* __restrict__ s, unsigned short* __restrict__ d, int n) {
  int i = blockIdx.x * blockDim.x + threadIdx.x;
  if (i < n) d[i] = f2bf(s[i]);
}

// Pack W[128,128] f32 (row-major K,N) into the WMMA-B register image:
//   Bp[((nt*4 + kstep)*32 + lane)*16 + e] = bf16( W[(kstep*32 + (lane>>4)*16 + e)*128 + nt*16 + (lane&15)] )
// Within a lane the 16 elements are K-consecutive, so e maps linearly to K.
__global__ void k_pack_B(const float* __restrict__ W, unsigned short* __restrict__ Bp) {
  int i = blockIdx.x * blockDim.x + threadIdx.x;
  if (i >= FDIM * FDIM) return;
  int e     = i & 15;
  int lane  = (i >> 4) & 31;
  int kstep = (i >> 9) & 3;
  int nt    = i >> 11;
  int k = kstep * 32 + (lane >> 4) * 16 + e;
  int n = nt * 16 + (lane & 15);
  Bp[i] = f2bf(W[k * FDIM + n]);
}

// C[M,128] = A[M,128] x B[128,128]; A bf16, B pre-packed bf16 (k_pack_B), C f32.
// block = 256 (8 waves); wave w handles columns [16w,16w+16).
// Each wave keeps its whole B column-tile (4 K-steps) in registers and sweeps
// MTPW M-tiles.  grid.x = NNODE/16/MTPW = 625.
__global__ void __launch_bounds__(256)
k_gemm_wmma(const unsigned short* __restrict__ A,
            const unsigned short* __restrict__ Bp,
            float* __restrict__ C) {
  const int lane  = threadIdx.x & 31;
  const int wave  = threadIdx.x >> 5;      // = N tile
  const int half  = lane >> 4;             // 0: lanes 0-15, 1: lanes 16-31
  const int l16   = lane & 15;
  const int koffA = half * 16;             // A: lanes16-31 hold K=16..31
  const int tileN = wave * 16;

  // load B: 4 K-steps x 32 contiguous bytes per lane (fully coalesced)
  ABu b[4];
  #pragma unroll
  for (int ks = 0; ks < 4; ++ks) {
    const uint4* bp = (const uint4*)(Bp + (((wave << 2) + ks) * 32 + lane) * 16);
    b[ks].q[0] = bp[0];
    b[ks].q[1] = bp[1];
  }

  #pragma unroll
  for (int t = 0; t < MTPW; ++t) {
    const int tileM = (blockIdx.x * MTPW + t) * 16;
    const int rowA  = tileM + l16;
    v8f acc = {};
    #pragma unroll
    for (int ks = 0; ks < 4; ++ks) {
      const int kb = ks * 32;
      ABu a;
      a.q[0] = *(const uint4*)(A + rowA * FDIM + kb + koffA);
      a.q[1] = *(const uint4*)(A + rowA * FDIM + kb + 8 + koffA);
      acc = __builtin_amdgcn_wmma_f32_16x16x32_bf16(
          false, a.v, false, b[ks].v, (short)0, acc, false, false);
    }
    // D 16x16 f32: VGPR r -> row r (+8 for lanes 16-31), col = lane&15
    const int rowC = tileM + half * 8;
    #pragma unroll
    for (int r = 0; r < 8; ++r)
      C[(rowC + r) * FDIM + tileN + l16] = acc[r];
  }
}

// el/er[n,h] = sum_d feat[n,h,d]*al/ar[h,d]
__global__ void k_attn_coef(const float* __restrict__ feat,
                            const float* __restrict__ al,
                            const float* __restrict__ ar,
                            float* __restrict__ el, float* __restrict__ er) {
  int i = blockIdx.x * blockDim.x + threadIdx.x;
  if (i >= NNODE * NHEAD) return;
  int node = i / NHEAD, h = i % NHEAD;
  const float* f = feat + node * FDIM + h * HDIM;
  const float* a = al + h * HDIM;
  const float* r = ar + h * HDIM;
  float sl = 0.f, sr = 0.f;
  #pragma unroll
  for (int d = 0; d < HDIM; ++d) { float fv = f[d]; sl += fv * a[d]; sr += fv * r[d]; }
  el[i] = sl; er[i] = sr;
}

__device__ __forceinline__ void atomicMaxF(float* addr, float val) {
  if (val >= 0.f) atomicMax((int*)addr, __float_as_int(val));
  else            atomicMin((unsigned int*)addr, __float_as_uint(val));
}

__device__ __forceinline__ float lrelu(float v) { return v > 0.f ? v : SLOPE * v; }

__global__ void k_edge_max(const float* __restrict__ el, const float* __restrict__ er,
                           const int* __restrict__ src, const int* __restrict__ dst,
                           float* __restrict__ emax) {
  int i = blockIdx.x * blockDim.x + threadIdx.x;
  if (i >= NEDGE * NHEAD) return;
  int e = i / NHEAD, h = i % NHEAD;
  int s = src[e], d = dst[e];
  float v = lrelu(el[s * NHEAD + h] + er[d * NHEAD + h]);
  atomicMaxF(&emax[d * NHEAD + h], v);
}

__global__ void k_fix_max(float* __restrict__ emax, int n) {
  int i = blockIdx.x * blockDim.x + threadIdx.x;
  if (i < n) { float v = emax[i]; if (!isfinite(v)) emax[i] = 0.f; }
}

__global__ void k_edge_expsum(const float* __restrict__ el, const float* __restrict__ er,
                              const float* __restrict__ emax,
                              const int* __restrict__ src, const int* __restrict__ dst,
                              float* __restrict__ wbuf, float* __restrict__ denom) {
  int i = blockIdx.x * blockDim.x + threadIdx.x;
  if (i >= NEDGE * NHEAD) return;
  int e = i / NHEAD, h = i % NHEAD;
  int s = src[e], d = dst[e];
  float v = lrelu(el[s * NHEAD + h] + er[d * NHEAD + h]);
  float w = __expf(v - emax[d * NHEAD + h]);
  wbuf[i] = w;
  atomicAdd(&denom[d * NHEAD + h], w);
}

__global__ void k_edge_aggr(const float* __restrict__ wbuf, const float* __restrict__ denom,
                            const float* __restrict__ feat,
                            const int* __restrict__ src, const int* __restrict__ dst,
                            float* __restrict__ rst) {
  int i = blockIdx.x * blockDim.x + threadIdx.x;
  if (i >= NEDGE * NHEAD) return;
  int e = i / NHEAD, h = i % NHEAD;
  int s = src[e], d = dst[e];
  float alpha = wbuf[i] / fmaxf(denom[d * NHEAD + h], 1e-9f);
  const float* f = feat + s * FDIM + h * HDIM;
  float*       o = rst  + d * FDIM + h * HDIM;
  #pragma unroll
  for (int c = 0; c < HDIM; ++c) atomicAdd(&o[c], f[c] * alpha);
}

// layer-1 epilogue: h1 = elu(rst + x + b); also emit bf16 for layer-2 GEMM
__global__ void k_epi1(const float* __restrict__ rst, const float* __restrict__ xin,
                       const float* __restrict__ b,
                       float* __restrict__ h1, unsigned short* __restrict__ hb) {
  int i = blockIdx.x * blockDim.x + threadIdx.x;
  if (i >= NNODE * FDIM) return;
  int c = i % FDIM;
  float v = rst[i] + xin[i] + b[c];
  v = v > 0.f ? v : (__expf(v) - 1.f);   // ELU(alpha=1)
  h1[i] = v;
  hb[i] = f2bf(v);
}

// layer-2 epilogue: out[n,c] = mean_h(rst[n,h,c] + h1[n,h,c] + b[h,c])
__global__ void k_epi2(const float* __restrict__ rst, const float* __restrict__ h1,
                       const float* __restrict__ b, float* __restrict__ out) {
  int i = blockIdx.x * blockDim.x + threadIdx.x;
  if (i >= NNODE * HDIM) return;
  int node = i / HDIM, c = i % HDIM;
  float s = 0.f;
  #pragma unroll
  for (int h = 0; h < NHEAD; ++h) {
    int idx = node * FDIM + h * HDIM + c;
    s += rst[idx] + h1[idx] + b[h * HDIM + c];
  }
  out[i] = s * (1.f / NHEAD);
}

static inline int nblk(long n) { return (int)((n + 255) / 256); }

extern "C" void kernel_launch(void* const* d_in, const int* in_sizes, int n_in,
                              void* d_out, int out_size, void* d_ws, size_t ws_size,
                              hipStream_t stream) {
  const float* x   = (const float*)d_in[0];
  const float* W1  = (const float*)d_in[1];
  const float* al1 = (const float*)d_in[2];
  const float* ar1 = (const float*)d_in[3];
  const float* b1  = (const float*)d_in[4];
  const float* W2  = (const float*)d_in[5];
  const float* al2 = (const float*)d_in[6];
  const float* ar2 = (const float*)d_in[7];
  const float* b2  = (const float*)d_in[8];
  const int*   src = (const int*)d_in[9];
  const int*   dst = (const int*)d_in[10];
  float* out = (float*)d_out;

  size_t off = 0;
  auto alloc = [&](size_t bytes) -> char* {
    char* p = (char*)d_ws + off;
    off += (bytes + 255) & ~(size_t)255;
    return p;
  };
  unsigned short* xb   = (unsigned short*)alloc((size_t)NNODE * FDIM * 2);
  unsigned short* wb   = (unsigned short*)alloc((size_t)FDIM * FDIM * 2);
  float* feat  = (float*)alloc((size_t)NNODE * FDIM * 4);
  float* h1    = (float*)alloc((size_t)NNODE * FDIM * 4);
  float* rst   = (float*)alloc((size_t)NNODE * FDIM * 4);
  float* el    = (float*)alloc((size_t)NNODE * NHEAD * 4);
  float* er    = (float*)alloc((size_t)NNODE * NHEAD * 4);
  float* emax  = (float*)alloc((size_t)NNODE * NHEAD * 4);
  float* denom = (float*)alloc((size_t)NNODE * NHEAD * 4);
  float* wbuf  = (float*)alloc((size_t)NEDGE * NHEAD * 4);

  const int T = 256;
  const int GEMM_GRID = NNODE / 16 / MTPW;   // 625
  const float NEG_INF = -__builtin_huge_valf();

  // ---------------- layer 1 ----------------
  k_conv_bf16<<<nblk((long)NNODE * FDIM), T, 0, stream>>>(x, xb, NNODE * FDIM);
  k_pack_B<<<nblk((long)FDIM * FDIM), T, 0, stream>>>(W1, wb);
  k_gemm_wmma<<<GEMM_GRID, 256, 0, stream>>>(xb, wb, feat);
  k_attn_coef<<<nblk((long)NNODE * NHEAD), T, 0, stream>>>(feat, al1, ar1, el, er);
  k_fill<<<nblk((long)NNODE * NHEAD), T, 0, stream>>>(emax, NEG_INF, NNODE * NHEAD);
  k_fill<<<nblk((long)NNODE * NHEAD), T, 0, stream>>>(denom, 0.f, NNODE * NHEAD);
  k_fill<<<nblk((long)NNODE * FDIM), T, 0, stream>>>(rst, 0.f, NNODE * FDIM);
  k_edge_max<<<nblk((long)NEDGE * NHEAD), T, 0, stream>>>(el, er, src, dst, emax);
  k_fix_max<<<nblk((long)NNODE * NHEAD), T, 0, stream>>>(emax, NNODE * NHEAD);
  k_edge_expsum<<<nblk((long)NEDGE * NHEAD), T, 0, stream>>>(el, er, emax, src, dst, wbuf, denom);
  k_edge_aggr<<<nblk((long)NEDGE * NHEAD), T, 0, stream>>>(wbuf, denom, feat, src, dst, rst);
  k_epi1<<<nblk((long)NNODE * FDIM), T, 0, stream>>>(rst, x, b1, h1, xb);

  // ---------------- layer 2 ----------------
  k_pack_B<<<nblk((long)FDIM * FDIM), T, 0, stream>>>(W2, wb);
  k_gemm_wmma<<<GEMM_GRID, 256, 0, stream>>>(xb, wb, feat);
  k_attn_coef<<<nblk((long)NNODE * NHEAD), T, 0, stream>>>(feat, al2, ar2, el, er);
  k_fill<<<nblk((long)NNODE * NHEAD), T, 0, stream>>>(emax, NEG_INF, NNODE * NHEAD);
  k_fill<<<nblk((long)NNODE * NHEAD), T, 0, stream>>>(denom, 0.f, NNODE * NHEAD);
  k_fill<<<nblk((long)NNODE * FDIM), T, 0, stream>>>(rst, 0.f, NNODE * FDIM);
  k_edge_max<<<nblk((long)NEDGE * NHEAD), T, 0, stream>>>(el, er, src, dst, emax);
  k_fix_max<<<nblk((long)NNODE * NHEAD), T, 0, stream>>>(emax, NNODE * NHEAD);
  k_edge_expsum<<<nblk((long)NEDGE * NHEAD), T, 0, stream>>>(el, er, emax, src, dst, wbuf, denom);
  k_edge_aggr<<<nblk((long)NEDGE * NHEAD), T, 0, stream>>>(wbuf, denom, feat, src, dst, rst);
  k_epi2<<<nblk((long)NNODE * HDIM), T, 0, stream>>>(rst, h1, b2, out);
}
